// EnhancedJointGraphPredictor_20392504721605
// MI455X (gfx1250) — matrix operand order repaired and checked by reading, not compiled, not run on hardware
//
#include <hip/hip_runtime.h>

#define DD 256
#define HH 4
#define CC 64
#define BB 1024
#define NEGS 0.2f

typedef __bf16 v16bf __attribute__((ext_vector_type(16)));
typedef float  v8f   __attribute__((ext_vector_type(8)));
typedef unsigned short u16x8  __attribute__((ext_vector_type(8)));
typedef unsigned short u16x16 __attribute__((ext_vector_type(16)));

__device__ __forceinline__ unsigned short f2bf(float f) {
  unsigned int u = __float_as_uint(f);
  u += 0x7FFFu + ((u >> 16) & 1u);          // round-to-nearest-even
  return (unsigned short)(u >> 16);
}
__device__ __forceinline__ float bf2f(unsigned short b) {
  return __uint_as_float(((unsigned int)b) << 16);
}
// float atomic max via integer atomics: monotone int trick (init slot to 0xFF800000 = -inf)
__device__ __forceinline__ void atomicMaxF(float* a, float v) {
  int iv = __float_as_int(v);
  if (iv >= 0) atomicMax((int*)a, iv);
  else         atomicMin((unsigned int*)a, (unsigned int)iv);
}

// ---------------------------------------------------------------------------
// WMMA bf16 GEMM: out = act(A[N,K]bf16 @ Wt[Dout,K]bf16^T + bias)
// One wave -> one 16x16 tile. 8 waves/block -> 128 output cols per block.
// ---------------------------------------------------------------------------
__global__ void gemm_wmma_bf16(const unsigned short* __restrict__ A, int lda,
                               const unsigned short* __restrict__ Wt,
                               const float* __restrict__ bias,
                               float* __restrict__ outF, int ldf,
                               unsigned short* __restrict__ outB, int ldb,
                               int Nrows, int K, int Dout, int relu) {
  const int wave  = threadIdx.x >> 5;
  const int lane  = threadIdx.x & 31;
  const int tilesN = Dout >> 4;
  const int tileN  = blockIdx.x * 8 + wave;
  if (tileN >= tilesN) return;              // wave-uniform exit, EXEC stays full
  const int l15 = lane & 15;
  const int hi  = lane >> 4;
  const int row0 = blockIdx.y << 4;
  int arow = row0 + l15; if (arow >= Nrows) arow = Nrows - 1;   // clamp, keep EXEC=all 1s
  const int col = (tileN << 4) + l15;
  const unsigned short* ap = A  + (size_t)arow * lda + (hi << 4);
  const unsigned short* bp = Wt + (size_t)col  * K   + (hi << 4);
  v8f acc = {};
  for (int k0 = 0; k0 < K; k0 += 32) {
    // A frag: lanes0-15 K=k0..k0+15, lanes16-31 K=k0+16..k0+31 (contiguous, matches B)
    u16x8  a0 = *(const u16x8*)(ap + k0);
    u16x8  a1 = *(const u16x8*)(ap + k0 + 8);
    // B frag: lanes0-15 K=k0..k0+15, lanes16-31 K=k0+16..k0+31 (contiguous in Wt row)
    u16x16 bu = *(const u16x16*)(bp + k0);
    u16x16 au;
#pragma unroll
    for (int i = 0; i < 8; ++i) { au[i] = a0[i]; au[i + 8] = a1[i]; }
    union U { u16x16 u; v16bf v; };
    U ua; ua.u = au;
    U ub; ub.u = bu;
    acc = __builtin_amdgcn_wmma_f32_16x16x32_bf16(false, ua.v, false, ub.v,
                                                  (short)0, acc, false, false);
  }
  const float bc = bias ? bias[col] : 0.0f;
#pragma unroll
  for (int i = 0; i < 8; ++i) {            // C/D layout: VGPR i -> M = hi*8 + i, N = l15
    int r = row0 + (hi << 3) + i;
    if (r >= Nrows) continue;
    float v = acc[i] + bc;
    if (relu) v = fmaxf(v, 0.0f);
    if (outF) outF[(size_t)r * ldf + col] = v;
    if (outB) outB[(size_t)r * ldb + col] = f2bf(v);
  }
}

// ---------------------------------------------------------------------------
// Helpers / elementwise kernels
// ---------------------------------------------------------------------------
__global__ void fill_u32(unsigned int* p, unsigned int v, long long n) {
  long long i = (long long)blockIdx.x * blockDim.x + threadIdx.x;
  if (i < n) p[i] = v;
}

// Wt[dout, k] = bf16(W[k, colOff+dout]); zero-pad K to Kpad
__global__ void cvt_wT(const float* __restrict__ W, int Kreal, int Kpad,
                       int DoutTot, int colOff, int Dout,
                       unsigned short* __restrict__ Wt) {
  long long t = (long long)blockIdx.x * blockDim.x + threadIdx.x;
  if (t >= (long long)Dout * Kpad) return;
  int dout = (int)(t / Kpad), k = (int)(t % Kpad);
  float v = (k < Kreal) ? W[(size_t)k * DoutTot + colOff + dout] : 0.0f;
  Wt[(size_t)dout * Kpad + k] = f2bf(v);
}

__global__ void cvt_pad_x(const float* __restrict__ x, int N, int Din,
                          unsigned short* __restrict__ xp) {
  long long t = (long long)blockIdx.x * blockDim.x + threadIdx.x;
  if (t >= (long long)N * 32) return;
  int n = (int)(t >> 5), c = (int)(t & 31);
  xp[t] = (c < Din) ? f2bf(x[(size_t)n * Din + c]) : (unsigned short)0;
}

__global__ void deg_kernel(const int* __restrict__ ei, int E, int N, float* deg) {
  int e = blockIdx.x * blockDim.x + threadIdx.x;
  if (e >= E + N) return;
  int dst = (e < E) ? ei[E + e] : (e - E);   // self loops appended
  atomicAdd(&deg[dst], 1.0f);
}
__global__ void dinv_kernel(const float* deg, float* dinv, int N) {
  int n = blockIdx.x * blockDim.x + threadIdx.x;
  if (n < N) dinv[n] = rsqrtf(fmaxf(deg[n], 1.0f));
}

// acc[dst, c..c+3] += h[src, c..c+3] * dinv[src]*dinv[dst]
__global__ void gcn_scatter(const int* __restrict__ ei, int E, int N,
                            const float* __restrict__ dinv,
                            const float* __restrict__ h, float* __restrict__ acc) {
  long long t = (long long)blockIdx.x * blockDim.x + threadIdx.x;
  if (t >= (long long)(E + N) * 64) return;
  int e = (int)(t >> 6), c = (int)((t & 63) << 2);
  int src, dst;
  if (e < E) { src = ei[e]; dst = ei[E + e]; } else { src = dst = e - E; }
  float norm = dinv[src] * dinv[dst];
  float4 hv = *(const float4*)(h + (size_t)src * DD + c);
  float* ap = acc + (size_t)dst * DD + c;
  atomicAdd(ap + 0, hv.x * norm); atomicAdd(ap + 1, hv.y * norm);
  atomicAdd(ap + 2, hv.z * norm); atomicAdd(ap + 3, hv.w * norm);
}

__global__ void gcn_finalize(const float* __restrict__ acc, const float* __restrict__ bias,
                             int N, unsigned short* __restrict__ outB,
                             float* __restrict__ outF) {
  long long t = (long long)blockIdx.x * blockDim.x + threadIdx.x;
  if (t >= (long long)N * DD) return;
  int c = (int)(t & 255);
  float v = fmaxf(acc[t] + bias[c], 0.0f);
  outB[t] = f2bf(v);
  if (outF) outF[t] = v;
}

__global__ void gat_sd(const float* __restrict__ h, const float* __restrict__ as,
                       const float* __restrict__ ad, float* __restrict__ s,
                       float* __restrict__ d, int N) {
  int t = blockIdx.x * blockDim.x + threadIdx.x;
  if (t >= N * HH) return;
  int n = t >> 2, hh = t & 3;
  const float* hp  = h  + (size_t)n * DD + hh * CC;
  const float* asp = as + hh * CC;
  const float* adp = ad + hh * CC;
  float ss = 0.f, dd = 0.f;
  for (int c = 0; c < CC; ++c) { float x = hp[c]; ss += x * asp[c]; dd += x * adp[c]; }
  s[t] = ss; d[t] = dd;
}

__global__ void gat_edge_max(const int* __restrict__ ei, int E, int N,
                             const float* __restrict__ s, const float* __restrict__ d,
                             float* __restrict__ m) {
  int t = blockIdx.x * blockDim.x + threadIdx.x;
  if (t >= (E + N) * HH) return;
  int e = t >> 2, hh = t & 3;
  int src, dst;
  if (e < E) { src = ei[e]; dst = ei[E + e]; } else { src = dst = e - E; }
  float v = s[src * 4 + hh] + d[dst * 4 + hh];
  v = (v > 0.f) ? v : NEGS * v;
  atomicMaxF(&m[dst * 4 + hh], v);
}

__global__ void gat_edge_accum(const int* __restrict__ ei, int E, int N,
                               const float* __restrict__ s, const float* __restrict__ d,
                               const float* __restrict__ m, const float* __restrict__ h,
                               float* __restrict__ sum, float* __restrict__ acc) {
  long long t = (long long)blockIdx.x * blockDim.x + threadIdx.x;
  if (t >= (long long)(E + N) * HH * 4) return;
  int chunk = (int)(t & 3), hh = (int)((t >> 2) & 3), e = (int)(t >> 4);
  int src, dst;
  if (e < E) { src = ei[e]; dst = ei[E + e]; } else { src = dst = e - E; }
  float v = s[src * 4 + hh] + d[dst * 4 + hh];
  v = (v > 0.f) ? v : NEGS * v;
  float ex = expf(v - m[dst * 4 + hh]);
  if (chunk == 0) atomicAdd(&sum[dst * 4 + hh], ex);
  const float* hp = h   + (size_t)src * DD + hh * CC + chunk * 16;
  float*       ap = acc + (size_t)dst * DD + hh * CC + chunk * 16;
#pragma unroll
  for (int j = 0; j < 16; j += 4) {
    float4 hv = *(const float4*)(hp + j);
    atomicAdd(ap + j + 0, ex * hv.x); atomicAdd(ap + j + 1, ex * hv.y);
    atomicAdd(ap + j + 2, ex * hv.z); atomicAdd(ap + j + 3, ex * hv.w);
  }
}

__global__ void gat_finalize(const float* __restrict__ acc, const float* __restrict__ sum,
                             const float* __restrict__ bias, int N,
                             unsigned short* __restrict__ outB, float* __restrict__ outF) {
  long long t = (long long)blockIdx.x * blockDim.x + threadIdx.x;
  if (t >= (long long)N * DD) return;
  int n = (int)(t >> 8), c = (int)(t & 255), hh = c >> 6;
  float v = acc[t] / (sum[n * 4 + hh] + 1e-16f) + bias[c];
  v = fmaxf(v, 0.0f);
  outB[t] = f2bf(v);
  if (outF) outF[t] = v;
}

__global__ void pool_acc(const float* __restrict__ x, const int* __restrict__ batch,
                         int N, float* __restrict__ s, float* __restrict__ mx) {
  long long t = (long long)blockIdx.x * blockDim.x + threadIdx.x;
  if (t >= (long long)N * DD) return;
  int n = (int)(t >> 8), c = (int)(t & 255);
  int b = batch[n];
  float v = x[t];
  atomicAdd(&s[(size_t)b * DD + c], v);
  atomicMaxF(&mx[(size_t)b * DD + c], v);
}
__global__ void pool_cnt(const int* __restrict__ batch, int N, float* cnt) {
  int n = blockIdx.x * blockDim.x + threadIdx.x;
  if (n < N) atomicAdd(&cnt[batch[n]], 1.0f);
}
__global__ void pool_finalize(const float* __restrict__ s, const float* __restrict__ mx,
                              const float* __restrict__ cnt,
                              unsigned short* __restrict__ mean_bf,
                              unsigned short* __restrict__ max_bf,
                              unsigned short* __restrict__ add_bf,
                              unsigned short* __restrict__ jfcat) {
  long long t = (long long)blockIdx.x * blockDim.x + threadIdx.x;
  if (t >= (long long)BB * DD) return;
  int b = (int)(t >> 8), c = (int)(t & 255);
  float cv = fmaxf(cnt[b], 1.0f);
  float mean = s[t] / cv;
  mean_bf[t] = f2bf(mean);
  jfcat[(size_t)b * 768 + 512 + c] = f2bf(mean);
  max_bf[t] = f2bf(mx[t]);
  add_bf[t] = f2bf(s[t]);
}

__global__ void prot_sum(const float* __restrict__ p, int N, float* g) {
  long long t = (long long)blockIdx.x * blockDim.x + threadIdx.x;
  if (t >= (long long)N * DD) return;
  atomicAdd(&g[t & 255], p[t]);
}
__global__ void prot_bcast(const float* __restrict__ g, float invN,
                           unsigned short* __restrict__ pg_bf,
                           unsigned short* __restrict__ jfcat) {
  long long t = (long long)blockIdx.x * blockDim.x + threadIdx.x;
  if (t >= (long long)BB * DD) return;
  int b = (int)(t >> 8), c = (int)(t & 255);
  unsigned short bf = f2bf(g[c] * invN);
  pg_bf[t] = bf;
  jfcat[(size_t)b * 768 + 256 + c] = bf;
}

__global__ void final_head(const unsigned short* __restrict__ h3,
                           const float* __restrict__ w, const float* __restrict__ b,
                           float* __restrict__ out) {
  int i = blockIdx.x * blockDim.x + threadIdx.x;
  if (i >= BB) return;
  float s = b[0];
  for (int k = 0; k < 64; ++k) s += bf2f(h3[i * 64 + k]) * w[k];
  out[i] = 1.0f / (1.0f + expf(-s));
}

// ---------------------------------------------------------------------------
extern "C" void kernel_launch(void* const* d_in, const int* in_sizes, int n_in,
                              void* d_out, int out_size, void* d_ws, size_t ws_size,
                              hipStream_t stream) {
  const float* mol_x  = (const float*)d_in[0];
  const int*   mei    = (const int*)d_in[1];
  const int*   mbatch = (const int*)d_in[2];
  const float* prot_x = (const float*)d_in[3];
  const int*   pei    = (const int*)d_in[4];
  const float *mg1_w=(const float*)d_in[5],  *mg1_b=(const float*)d_in[6];
  const float *mg2_w=(const float*)d_in[7],  *mg2_b=(const float*)d_in[8];
  const float *mg3_w=(const float*)d_in[9],  *mg3_b=(const float*)d_in[10];
  const float *pg1_w=(const float*)d_in[11], *pg1_b=(const float*)d_in[12];
  const float *pg2_w=(const float*)d_in[13], *pg2_b=(const float*)d_in[14];
  const float *ma1_w=(const float*)d_in[15], *ma1_as=(const float*)d_in[16],
              *ma1_ad=(const float*)d_in[17], *ma1_b=(const float*)d_in[18];
  const float *ma2_w=(const float*)d_in[19], *ma2_as=(const float*)d_in[20],
              *ma2_ad=(const float*)d_in[21], *ma2_b=(const float*)d_in[22];
  const float *pa_w=(const float*)d_in[23],  *pa_as=(const float*)d_in[24],
              *pa_ad=(const float*)d_in[25],  *pa_b=(const float*)d_in[26];
  const float *qkv_w=(const float*)d_in[27], *qkv_b=(const float*)d_in[28];
  const float *att_w=(const float*)d_in[29], *att_b=(const float*)d_in[30];
  const float *jt_w1=(const float*)d_in[31], *jt_b1=(const float*)d_in[32];
  const float *jt_w2=(const float*)d_in[33], *jt_b2=(const float*)d_in[34];
  const float *pl0_w=(const float*)d_in[35], *pl0_b=(const float*)d_in[36];
  const float *pl1_w=(const float*)d_in[37], *pl1_b=(const float*)d_in[38];
  const float *pl2_w=(const float*)d_in[39], *pl2_b=(const float*)d_in[40];
  const float *pr1_w=(const float*)d_in[41], *pr1_b=(const float*)d_in[42];
  const float *pr2_w=(const float*)d_in[43], *pr2_b=(const float*)d_in[44];
  const float *pr3_w=(const float*)d_in[45], *pr3_b=(const float*)d_in[46];
  const float *pr4_w=(const float*)d_in[47], *pr4_b=(const float*)d_in[48];

  const int Nm = in_sizes[0] / 8;
  const int Em = in_sizes[1] / 2;
  const int Np = in_sizes[3] / 5;
  const int Ep = in_sizes[4] / 2;

  // ---- workspace bump allocator (256B aligned) ----
  char* base = (char*)d_ws; size_t off = 0;
  auto alloc = [&](size_t bytes) -> void* {
    off = (off + 255) & ~(size_t)255;
    void* p = base + off; off += bytes; return p;
  };
  // weights (bf16, transposed to [Dout, Kpad])
  unsigned short* wt_mg1 = (unsigned short*)alloc(256 * 32 * 2);
  unsigned short* wt_mg2 = (unsigned short*)alloc(256 * 256 * 2);
  unsigned short* wt_mg3 = (unsigned short*)alloc(256 * 256 * 2);
  unsigned short* wt_ma1 = (unsigned short*)alloc(256 * 256 * 2);
  unsigned short* wt_ma2 = (unsigned short*)alloc(256 * 256 * 2);
  unsigned short* wt_pg1 = (unsigned short*)alloc(256 * 32 * 2);
  unsigned short* wt_pg2 = (unsigned short*)alloc(256 * 256 * 2);
  unsigned short* wt_pa  = (unsigned short*)alloc(256 * 256 * 2);
  unsigned short* wt_v   = (unsigned short*)alloc(256 * 256 * 2);
  unsigned short* wt_att = (unsigned short*)alloc(256 * 256 * 2);
  unsigned short* wt_jt1 = (unsigned short*)alloc(512 * 768 * 2);
  unsigned short* wt_jt2 = (unsigned short*)alloc(256 * 512 * 2);
  unsigned short* wt_pl0 = (unsigned short*)alloc(128 * 256 * 2);
  unsigned short* wt_pl1 = (unsigned short*)alloc(128 * 256 * 2);
  unsigned short* wt_pl2 = (unsigned short*)alloc(128 * 256 * 2);
  unsigned short* wt_pr1 = (unsigned short*)alloc(256 * 640 * 2);
  unsigned short* wt_pr2 = (unsigned short*)alloc(128 * 256 * 2);
  unsigned short* wt_pr3 = (unsigned short*)alloc(64 * 128 * 2);
  // towers
  unsigned short* xpad_m = (unsigned short*)alloc((size_t)Nm * 32 * 2);
  unsigned short* xpad_p = (unsigned short*)alloc((size_t)Np * 32 * 2);
  float* hbuf_m = (float*)alloc((size_t)Nm * DD * 4);
  float* acc_m  = (float*)alloc((size_t)Nm * DD * 4);
  unsigned short* xbf_m = (unsigned short*)alloc((size_t)Nm * DD * 2);
  float* hbuf_p = (float*)alloc((size_t)Np * DD * 4);
  float* acc_p  = (float*)alloc((size_t)Np * DD * 4);
  unsigned short* xbf_p = (unsigned short*)alloc((size_t)Np * DD * 2);
  float* deg_m  = (float*)alloc((size_t)Nm * 4);
  float* dinv_m = (float*)alloc((size_t)Nm * 4);
  float* deg_p  = (float*)alloc((size_t)Np * 4);
  float* dinv_p = (float*)alloc((size_t)Np * 4);
  float* sS   = (float*)alloc((size_t)Nm * HH * 4);   // reused by prot (Np<=Nm)
  float* sDv  = (float*)alloc((size_t)Nm * HH * 4);
  float* sM   = (float*)alloc((size_t)Nm * HH * 4);
  float* sSum = (float*)alloc((size_t)Nm * HH * 4);
  // pooling / head
  float* pool_s = (float*)alloc((size_t)BB * DD * 4);
  float* pool_m = (float*)alloc((size_t)BB * DD * 4);
  float* cnt    = (float*)alloc((size_t)BB * 4);
  float* pgsum  = (float*)alloc(DD * 4);
  unsigned short* pg_bf   = (unsigned short*)alloc((size_t)BB * DD * 2);
  unsigned short* mean_bf = (unsigned short*)alloc((size_t)BB * DD * 2);
  unsigned short* max_bf  = (unsigned short*)alloc((size_t)BB * DD * 2);
  unsigned short* add_bf  = (unsigned short*)alloc((size_t)BB * DD * 2);
  unsigned short* vb      = (unsigned short*)alloc((size_t)BB * DD * 2);
  unsigned short* jfcat   = (unsigned short*)alloc((size_t)BB * 768 * 2);
  unsigned short* jt1b    = (unsigned short*)alloc((size_t)BB * 512 * 2);
  unsigned short* fcat    = (unsigned short*)alloc((size_t)BB * 640 * 2);
  unsigned short* h1b     = (unsigned short*)alloc((size_t)BB * 256 * 2);
  unsigned short* h2b     = (unsigned short*)alloc((size_t)BB * 128 * 2);
  unsigned short* h3b     = (unsigned short*)alloc((size_t)BB * 64 * 2);
  (void)ws_size; (void)n_in; (void)out_size;

  auto nb = [](long long n) { return (unsigned)((n + 255) / 256); };
  auto cvtW = [&](const float* W, int Kreal, int Kpad, int DoutTot, int colOff,
                  int Dout, unsigned short* Wt) {
    cvt_wT<<<nb((long long)Dout * Kpad), 256, 0, stream>>>(W, Kreal, Kpad, DoutTot,
                                                           colOff, Dout, Wt);
  };
  auto gemm = [&](const unsigned short* A, int lda, const unsigned short* Wt,
                  const float* bias, float* outF, int ldf, unsigned short* outB,
                  int ldb, int Nrows, int K, int Dout, int relu) {
    dim3 g((unsigned)((Dout / 16 + 7) / 8), (unsigned)((Nrows + 15) / 16));
    gemm_wmma_bf16<<<g, 256, 0, stream>>>(A, lda, Wt, bias, outF, ldf, outB, ldb,
                                          Nrows, K, Dout, relu);
  };

  // ---- weight convert/transpose ----
  cvtW(mg1_w, 8,   32,  256, 0,   256, wt_mg1);
  cvtW(mg2_w, 256, 256, 256, 0,   256, wt_mg2);
  cvtW(mg3_w, 256, 256, 256, 0,   256, wt_mg3);
  cvtW(ma1_w, 256, 256, 256, 0,   256, wt_ma1);
  cvtW(ma2_w, 256, 256, 256, 0,   256, wt_ma2);
  cvtW(pg1_w, 5,   32,  256, 0,   256, wt_pg1);
  cvtW(pg2_w, 256, 256, 256, 0,   256, wt_pg2);
  cvtW(pa_w,  256, 256, 256, 0,   256, wt_pa);
  cvtW(qkv_w, 256, 256, 768, 512, 256, wt_v);     // v-projection slice
  cvtW(att_w, 256, 256, 256, 0,   256, wt_att);
  cvtW(jt_w1, 768, 768, 512, 0,   512, wt_jt1);
  cvtW(jt_w2, 512, 512, 256, 0,   256, wt_jt2);
  cvtW(pl0_w, 256, 256, 128, 0,   128, wt_pl0);
  cvtW(pl1_w, 256, 256, 128, 0,   128, wt_pl1);
  cvtW(pl2_w, 256, 256, 128, 0,   128, wt_pl2);
  cvtW(pr1_w, 640, 640, 256, 0,   256, wt_pr1);
  cvtW(pr2_w, 256, 256, 128, 0,   128, wt_pr2);
  cvtW(pr3_w, 128, 128, 64,  0,   64,  wt_pr3);

  cvt_pad_x<<<nb((long long)Nm * 32), 256, 0, stream>>>(mol_x, Nm, 8, xpad_m);
  cvt_pad_x<<<nb((long long)Np * 32), 256, 0, stream>>>(prot_x, Np, 5, xpad_p);

  // ---- degrees (with self loops) ----
  fill_u32<<<nb(Nm), 256, 0, stream>>>((unsigned*)deg_m, 0u, Nm);
  deg_kernel<<<nb(Em + Nm), 256, 0, stream>>>(mei, Em, Nm, deg_m);
  dinv_kernel<<<nb(Nm), 256, 0, stream>>>(deg_m, dinv_m, Nm);
  fill_u32<<<nb(Np), 256, 0, stream>>>((unsigned*)deg_p, 0u, Np);
  deg_kernel<<<nb(Ep + Np), 256, 0, stream>>>(pei, Ep, Np, deg_p);
  dinv_kernel<<<nb(Np), 256, 0, stream>>>(deg_p, dinv_p, Np);

  auto run_gcn = [&](const unsigned short* xin, int lda, const unsigned short* wt,
                     int K, const float* bias, const int* ei, int E, int N,
                     const float* dinv, float* hbuf, float* accb,
                     unsigned short* xout, float* xoutF) {
    gemm(xin, lda, wt, nullptr, hbuf, DD, nullptr, 0, N, K, DD, 0);
    fill_u32<<<nb((long long)N * DD), 256, 0, stream>>>((unsigned*)accb, 0u,
                                                        (long long)N * DD);
    gcn_scatter<<<nb((long long)(E + N) * 64), 256, 0, stream>>>(ei, E, N, dinv,
                                                                 hbuf, accb);
    gcn_finalize<<<nb((long long)N * DD), 256, 0, stream>>>(accb, bias, N, xout,
                                                            xoutF);
  };
  auto run_gat = [&](const unsigned short* xin, const unsigned short* wt,
                     const float* as, const float* ad, const float* bias,
                     const int* ei, int E, int N, float* hbuf, float* accb,
                     unsigned short* xout, float* xoutF) {
    gemm(xin, DD, wt, nullptr, hbuf, DD, nullptr, 0, N, DD, DD, 0);
    gat_sd<<<nb((long long)N * HH), 256, 0, stream>>>(hbuf, as, ad, sS, sDv, N);
    fill_u32<<<nb((long long)N * HH), 256, 0, stream>>>((unsigned*)sM, 0xFF800000u,
                                                        (long long)N * HH);
    fill_u32<<<nb((long long)N * HH), 256, 0, stream>>>((unsigned*)sSum, 0u,
                                                        (long long)N * HH);
    fill_u32<<<nb((long long)N * DD), 256, 0, stream>>>((unsigned*)accb, 0u,
                                                        (long long)N * DD);
    gat_edge_max<<<nb((long long)(E + N) * HH), 256, 0, stream>>>(ei, E, N, sS, sDv, sM);
    gat_edge_accum<<<nb((long long)(E + N) * HH * 4), 256, 0, stream>>>(
        ei, E, N, sS, sDv, sM, hbuf, sSum, accb);
    gat_finalize<<<nb((long long)N * DD), 256, 0, stream>>>(accb, sSum, bias, N,
                                                            xout, xoutF);
  };

  // ---- molecule tower ----
  run_gcn(xpad_m, 32, wt_mg1, 32, mg1_b, mei, Em, Nm, dinv_m, hbuf_m, acc_m, xbf_m, nullptr);
  run_gcn(xbf_m, DD, wt_mg2, DD, mg2_b, mei, Em, Nm, dinv_m, hbuf_m, acc_m, xbf_m, nullptr);
  run_gcn(xbf_m, DD, wt_mg3, DD, mg3_b, mei, Em, Nm, dinv_m, hbuf_m, acc_m, xbf_m, nullptr);
  run_gat(xbf_m, wt_ma1, ma1_as, ma1_ad, ma1_b, mei, Em, Nm, hbuf_m, acc_m, xbf_m, nullptr);
  run_gat(xbf_m, wt_ma2, ma2_as, ma2_ad, ma2_b, mei, Em, Nm, hbuf_m, acc_m, xbf_m, hbuf_m);

  // ---- molecule pooling (x_final f32 in hbuf_m) ----
  fill_u32<<<nb((long long)BB * DD), 256, 0, stream>>>((unsigned*)pool_s, 0u, (long long)BB * DD);
  fill_u32<<<nb((long long)BB * DD), 256, 0, stream>>>((unsigned*)pool_m, 0xFF800000u, (long long)BB * DD);
  fill_u32<<<nb(BB), 256, 0, stream>>>((unsigned*)cnt, 0u, BB);
  pool_acc<<<nb((long long)Nm * DD), 256, 0, stream>>>(hbuf_m, mbatch, Nm, pool_s, pool_m);
  pool_cnt<<<nb(Nm), 256, 0, stream>>>(mbatch, Nm, cnt);

  // ---- protein tower ----
  run_gcn(xpad_p, 32, wt_pg1, 32, pg1_b, pei, Ep, Np, dinv_p, hbuf_p, acc_p, xbf_p, nullptr);
  run_gcn(xbf_p, DD, wt_pg2, DD, pg2_b, pei, Ep, Np, dinv_p, hbuf_p, acc_p, xbf_p, nullptr);
  run_gat(xbf_p, wt_pa, pa_as, pa_ad, pa_b, pei, Ep, Np, hbuf_p, acc_p, xbf_p, hbuf_p);

  fill_u32<<<nb(DD), 256, 0, stream>>>((unsigned*)pgsum, 0u, DD);
  prot_sum<<<nb((long long)Np * DD), 256, 0, stream>>>(hbuf_p, Np, pgsum);
  prot_bcast<<<nb((long long)BB * DD), 256, 0, stream>>>(pgsum, 1.0f / (float)Np, pg_bf, jfcat);
  pool_finalize<<<nb((long long)BB * DD), 256, 0, stream>>>(pool_s, pool_m, cnt,
                                                            mean_bf, max_bf, add_bf, jfcat);

  // ---- head ----
  gemm(pg_bf, DD, wt_v, qkv_b + 512, nullptr, 0, vb, DD, BB, DD, DD, 0);          // v
  gemm(vb, DD, wt_att, att_b, nullptr, 0, jfcat + 0, 768, BB, DD, DD, 0);          // attended -> jf[:,0:256]
  gemm(jfcat, 768, wt_jt1, jt_b1, nullptr, 0, jt1b, 512, BB, 768, 512, 1);         // jt1 relu
  gemm(jt1b, 512, wt_jt2, jt_b2, nullptr, 0, fcat + 0, 640, BB, 512, 256, 1);      // jt2 -> f[:,0:256]
  gemm(mean_bf, DD, wt_pl0, pl0_b, nullptr, 0, fcat + 256, 640, BB, DD, 128, 0);   // pm
  gemm(max_bf,  DD, wt_pl1, pl1_b, nullptr, 0, fcat + 384, 640, BB, DD, 128, 0);   // px
  gemm(add_bf,  DD, wt_pl2, pl2_b, nullptr, 0, fcat + 512, 640, BB, DD, 128, 0);   // pa
  gemm(fcat, 640, wt_pr1, pr1_b, nullptr, 0, h1b, 256, BB, 640, 256, 1);
  gemm(h1b, 256, wt_pr2, pr2_b, nullptr, 0, h2b, 128, BB, 256, 128, 1);
  gemm(h2b, 128, wt_pr3, pr3_b, nullptr, 0, h3b, 64, BB, 128, 64, 1);
  final_head<<<nb(BB), 256, 0, stream>>>(h3b, pr4_w, pr4_b, (float*)d_out);
}
